// Diff2StructLSTM_43465069035647
// MI455X (gfx1250) — matrix operand order repaired and checked by reading, compiled, not run on hardware
//
#include <hip/hip_runtime.h>
#include <hip/hip_bf16.h>
#include <math.h>

// ---------------------------------------------------------------------------
// Types for CDNA5 WMMA (gfx1250, wave32)
// ---------------------------------------------------------------------------
typedef __attribute__((ext_vector_type(16))) __bf16 v16bf;
typedef __attribute__((ext_vector_type(8)))  __bf16 v8bf;
typedef __attribute__((ext_vector_type(8)))  float  v8f;

#define P_PEAKS 2048
#define HID     768
#define G4      3072   // 4*HID

__device__ __forceinline__ float sigm(float x) {
    return 1.0f / (1.0f + __expf(-x));
}

// ---------------------------------------------------------------------------
// Kernel 0: weight convert + transpose:  dst[N,Kp] (bf16) = src[K,N] (f32)^T,
// zero-padded to Kp in K. One thread per destination element.
// ---------------------------------------------------------------------------
__global__ __launch_bounds__(256) void conv_wt_kernel(
    const float* __restrict__ src, __bf16* __restrict__ dst,
    int K, int Kp, int N)
{
    const int idx = blockIdx.x * blockDim.x + threadIdx.x;
    if (idx >= N * Kp) return;
    const int n = idx / Kp;
    const int k = idx - n * Kp;
    dst[idx] = (__bf16)((k < K) ? src[k * N + n] : 0.f);
}

// ---------------------------------------------------------------------------
// Kernel 1: per-peak embeddings.
// One wave per peak; layer-1 embed LSTM (3->64) distributed 2 units/lane with
// __shfl broadcast of h; layer-2 (64->6) + intensity MLP computed redundantly.
// Lane 0 writes the bf16 input row [32] = {I_embed(6), hkl_embed(6), 0...}.
// ---------------------------------------------------------------------------
__global__ __launch_bounds__(256) void embed_kernel(
    const float* __restrict__ Is, const float* __restrict__ hklss,
    const float* __restrict__ WI1, const float* __restrict__ bI1,
    const float* __restrict__ WI2, const float* __restrict__ bI2,
    const float* __restrict__ Wx1, const float* __restrict__ Wh1,
    const float* __restrict__ b1,
    const float* __restrict__ Wx2, const float* __restrict__ Wh2,
    const float* __restrict__ b2,
    __bf16* __restrict__ inp_bf /* [P,32] */)
{
    const int lane = threadIdx.x & 31;
    const int wid  = threadIdx.x >> 5;
    const int p    = blockIdx.x * (blockDim.x >> 5) + wid;
    if (p >= P_PEAKS) return;

    const int u0 = lane, u1 = lane + 32;

    float h0r = 0.f, h1r = 0.f, c0r = 0.f, c1r = 0.f;   // layer-1 state
    float hh[6], cc[6];                                  // layer-2 state
#pragma unroll
    for (int u = 0; u < 6; ++u) { hh[u] = 0.f; cc[u] = 0.f; }

    for (int t = 0; t < 8; ++t) {
        const float* xt = hklss + (p * 8 + t) * 3;
        const float x0 = xt[0], x1 = xt[1], x2 = xt[2];

        float acc[8];
#pragma unroll
        for (int j = 0; j < 8; ++j) {
            const int col = (j < 4) ? (j * 64 + u0) : ((j - 4) * 64 + u1);
            acc[j] = b1[col] + x0 * Wx1[col] + x1 * Wx1[256 + col]
                             + x2 * Wx1[512 + col];
        }
        for (int k = 0; k < 32; ++k) {
            const float hk = __shfl(h0r, k);
            const float* w = Wh1 + k * 256;
#pragma unroll
            for (int j = 0; j < 8; ++j) {
                const int col = (j < 4) ? (j * 64 + u0) : ((j - 4) * 64 + u1);
                acc[j] += hk * w[col];
            }
        }
        for (int k = 0; k < 32; ++k) {
            const float hk = __shfl(h1r, k);
            const float* w = Wh1 + (k + 32) * 256;
#pragma unroll
            for (int j = 0; j < 8; ++j) {
                const int col = (j < 4) ? (j * 64 + u0) : ((j - 4) * 64 + u1);
                acc[j] += hk * w[col];
            }
        }
        // gate order: o, f, i, g
        {
            const float o = sigm(acc[0]), f = sigm(acc[1]);
            const float ii = sigm(acc[2]), g = tanhf(acc[3]);
            c0r = f * c0r + ii * g;  h0r = o * tanhf(c0r);
        }
        {
            const float o = sigm(acc[4]), f = sigm(acc[5]);
            const float ii = sigm(acc[6]), g = tanhf(acc[7]);
            c1r = f * c1r + ii * g;  h1r = o * tanhf(c1r);
        }

        // ---- layer-2 LSTM step on relu(h1_t), redundant per lane ----
        float a2[24];
#pragma unroll
        for (int j = 0; j < 24; ++j) a2[j] = b2[j];
        for (int k = 0; k < 32; ++k) {
            const float rk = fmaxf(__shfl(h0r, k), 0.f);
            const float* w = Wx2 + k * 24;
#pragma unroll
            for (int j = 0; j < 24; ++j) a2[j] += rk * w[j];
        }
        for (int k = 0; k < 32; ++k) {
            const float rk = fmaxf(__shfl(h1r, k), 0.f);
            const float* w = Wx2 + (k + 32) * 24;
#pragma unroll
            for (int j = 0; j < 24; ++j) a2[j] += rk * w[j];
        }
#pragma unroll
        for (int k = 0; k < 6; ++k) {
            const float hk = hh[k];
            const float* w = Wh2 + k * 24;
#pragma unroll
            for (int j = 0; j < 24; ++j) a2[j] += hk * w[j];
        }
#pragma unroll
        for (int u = 0; u < 6; ++u) {
            const float o = sigm(a2[u]), f = sigm(a2[6 + u]);
            const float ii = sigm(a2[12 + u]), g = tanhf(a2[18 + u]);
            cc[u] = f * cc[u] + ii * g;
            hh[u] = o * tanhf(cc[u]);
        }
    }

    // ---- intensity MLP: 1 -> 64 (relu) -> 6 (relu) ----
    const float Iv = Is[p];
    float e[6];
#pragma unroll
    for (int j = 0; j < 6; ++j) e[j] = bI2[j];
    for (int k = 0; k < 64; ++k) {
        const float t1 = fmaxf(Iv * WI1[k] + bI1[k], 0.f);
        const float* w = WI2 + k * 6;
#pragma unroll
        for (int j = 0; j < 6; ++j) e[j] += t1 * w[j];
    }

    if (lane == 0) {
        __bf16* o = inp_bf + p * 32;
#pragma unroll
        for (int j = 0; j < 6; ++j) {
            o[j]     = (__bf16)fmaxf(e[j], 0.f);   // I_embed (relu)
            o[6 + j] = (__bf16)fmaxf(hh[j], 0.f);  // hkl_embed (relu)
        }
#pragma unroll
        for (int j = 12; j < 32; ++j) o[j] = (__bf16)0.f;   // K padding
    }
}

// ---------------------------------------------------------------------------
// Kernel 2: register-blocked WMMA GEMM.
//   C[M,N] = act( A[M,K] @ W[K,N] + b[N] )
//   A  : bf16 row-major [M,K]   (relu'd / K-padded by producer)
//   WT : bf16 [N,K] = W^T       (from conv_wt_kernel)
//   out: OUT_BF16_RELU ? bf16 relu(out) : f32 out
// One wave computes a 32x64 macro-tile (2 M-tiles x 4 N-tiles): each A
// fragment feeds 4 WMMAs, each B fragment feeds 2 -> 8 v_wmma per 12 b128
// loads. K is compile-time so all fragment loads are plain aligned vector
// loads. Fragment layouts used here (wave32, contiguous K, same order in A and B):
//   A: lane l row M=l&15; lanes 0-15 K=0..15, lanes 16-31
//      K=16..31 -> two contiguous 16B loads.
//   B: lane l col N=l&15; lanes 0-15 K=0..15, lanes 16-31 K=16..31
//      -> one contiguous 32B load from WT.
//   C/D: VGPR r -> M = r + 8*(l>=16), N = l&15.
// ---------------------------------------------------------------------------
template<int K, bool OUT_BF16_RELU>
__global__ __launch_bounds__(256) void wmma_gemm_kernel(
    const __bf16* __restrict__ A, const __bf16* __restrict__ WT,
    const float* __restrict__ bias, void* __restrict__ Cout,
    int M, int N)
{
    const int wave  = (blockIdx.x * blockDim.x + threadIdx.x) >> 5;
    const int lane  = threadIdx.x & 31;
    const int nblk  = N >> 6;            // 64-wide N macro-tiles
    const int bm    = wave / nblk;
    const int bn    = wave % nblk;
    if (bm * 32 >= M) return;

    const int half = lane >> 4;          // 0: lanes 0-15, 1: lanes 16-31
    const int rc   = lane & 15;

    const __bf16* arow0 = A + (size_t)(bm * 32 + rc) * K + half * 16;
    const __bf16* arow1 = arow0 + (size_t)16 * K;
    const __bf16* wrow[4];
#pragma unroll
    for (int q = 0; q < 4; ++q)
        wrow[q] = WT + (size_t)(bn * 64 + q * 16 + rc) * K + half * 16;

    v8f acc[2][4];
#pragma unroll
    for (int mi = 0; mi < 2; ++mi)
#pragma unroll
        for (int q = 0; q < 4; ++q) acc[mi][q] = (v8f){};

#pragma unroll 2
    for (int kb = 0; kb < K; kb += 32) {
        if (kb + 32 < K) {               // prefetch next K-chunk of weights
            __builtin_prefetch(wrow[0] + kb + 32, 0, 1);
            __builtin_prefetch(wrow[2] + kb + 32, 0, 1);
        }
        const v8bf a0lo = *(const v8bf*)(arow0 + kb);
        const v8bf a0hi = *(const v8bf*)(arow0 + kb + 8);
        const v8bf a1lo = *(const v8bf*)(arow1 + kb);
        const v8bf a1hi = *(const v8bf*)(arow1 + kb + 8);
        const v16bf af0 = __builtin_shufflevector(a0lo, a0hi,
            0, 1, 2, 3, 4, 5, 6, 7, 8, 9, 10, 11, 12, 13, 14, 15);
        const v16bf af1 = __builtin_shufflevector(a1lo, a1hi,
            0, 1, 2, 3, 4, 5, 6, 7, 8, 9, 10, 11, 12, 13, 14, 15);
#pragma unroll
        for (int q = 0; q < 4; ++q) {
            const v16bf bf = *(const v16bf*)(wrow[q] + kb);
            acc[0][q] = __builtin_amdgcn_wmma_f32_16x16x32_bf16(
                false, af0, false, bf, (short)0, acc[0][q], false, false);
            acc[1][q] = __builtin_amdgcn_wmma_f32_16x16x32_bf16(
                false, af1, false, bf, (short)0, acc[1][q], false, false);
        }
    }

#pragma unroll
    for (int q = 0; q < 4; ++q) {
        const int col = bn * 64 + q * 16 + rc;
        const float bn_v = bias[col];
#pragma unroll
        for (int mi = 0; mi < 2; ++mi) {
#pragma unroll
            for (int r = 0; r < 8; ++r) {
                const int m = bm * 32 + mi * 16 + r + half * 8;
                const float v = acc[mi][q][r] + bn_v;
                if (OUT_BF16_RELU) {
                    ((__bf16*)Cout)[(size_t)m * N + col] =
                        (__bf16)fmaxf(v, 0.f);
                } else {
                    ((float*)Cout)[(size_t)m * N + col] = v;
                }
            }
        }
    }
}

// ---------------------------------------------------------------------------
// Kernel 3: f-embedding  relu( relu(fs@W1 + b1) @ W2 + b2 )  -> [768] f32
// ---------------------------------------------------------------------------
__global__ __launch_bounds__(256) void femb_kernel(
    const float* __restrict__ fs,
    const float* __restrict__ W1, const float* __restrict__ b1,
    const float* __restrict__ W2, const float* __restrict__ b2,
    float* __restrict__ out768)
{
    __shared__ float t1[512];
    const int j = threadIdx.x;
    const float f0 = fs[0], f1 = fs[1];
#pragma unroll
    for (int r = 0; r < 2; ++r) {
        const int k = j + 256 * r;
        t1[k] = fmaxf(f0 * W1[k] + f1 * W1[512 + k] + b1[k], 0.f);
    }
    __syncthreads();
#pragma unroll
    for (int r = 0; r < 3; ++r) {
        const int colm = j + 256 * r;
        float acc = b2[colm];
        for (int k = 0; k < 512; ++k) acc += t1[k] * W2[k * HID + colm];
        out768[colm] = fmaxf(acc, 0.f);
    }
}

// ---------------------------------------------------------------------------
// Kernel 4: sequential main LSTM layer (latency-bound, single workgroup).
// h/c in LDS; f32 weights stream from L2. Gpre = relu(x)@Wxg + bg precomputed
// by the WMMA GEMM. Emits bf16 relu(h_t) rows (next layer's WMMA A operand)
// and the f32 final row for the output tanh.
//   g = Gpre[t] + h@Whg ; c' = sig(f)*c + sig(i)*tanh(g)
//   h' = sig(o) * tanh( h@Whr + c'@Wcr + br )      (uses PREVIOUS h)
// ---------------------------------------------------------------------------
__global__ __launch_bounds__(1024) void main_lstm_kernel(
    const float* __restrict__ G, const float* __restrict__ h0,
    const float* __restrict__ Whg, const float* __restrict__ Wcr,
    const float* __restrict__ Whr, const float* __restrict__ br,
    __bf16* __restrict__ xbf_out /* [T,HID] relu'd bf16 */,
    float* __restrict__ hlast    /* [HID] f32, t = T-1 */)
{
    __shared__ float h_s[HID];
    __shared__ float c_s[HID];
    __shared__ float o_s[HID];
    __shared__ float gacc[G4];

    const int j = threadIdx.x;             // 0..1023
    if (j < HID) { h_s[j] = h0[j]; c_s[j] = 0.f; }
    __syncthreads();

    for (int t = 0; t < P_PEAKS; ++t) {
        // phase 1: gate pre-activations (all 1024 threads, 3 columns each)
        const float* Gt = G + (size_t)t * G4;
        float a0 = Gt[j], a1 = Gt[j + 1024], a2 = Gt[j + 2048];
#pragma unroll 4
        for (int k = 0; k < HID; ++k) {
            const float hk = h_s[k];                 // LDS broadcast
            const float* w = Whg + (size_t)k * G4 + j;
            a0 += hk * w[0];
            a1 += hk * w[1024];
            a2 += hk * w[2048];
        }
        gacc[j] = a0; gacc[j + 1024] = a1; gacc[j + 2048] = a2;
        __syncthreads();

        // phase 2: cell update (gate order o, f, i, g)
        if (j < HID) {
            const float o  = sigm(gacc[j]);
            const float f  = sigm(gacc[HID + j]);
            const float ii = sigm(gacc[2 * HID + j]);
            const float g  = tanhf(gacc[3 * HID + j]);
            c_s[j] = f * c_s[j] + ii * g;
            o_s[j] = o;
        }
        __syncthreads();

        // phase 3: recurrent projection for h' (reads OLD h_s, NEW c_s)
        float hn = 0.f;
        if (j < HID) {
            float acc = br[j];
#pragma unroll 4
            for (int k = 0; k < HID; ++k) {
                acc += h_s[k] * Whr[(size_t)k * HID + j]
                     + c_s[k] * Wcr[(size_t)k * HID + j];
            }
            hn = o_s[j] * tanhf(acc);
            xbf_out[(size_t)t * HID + j] = (__bf16)fmaxf(hn, 0.f);
            if (t == P_PEAKS - 1) hlast[j] = hn;
        }
        __syncthreads();
        if (j < HID) h_s[j] = hn;
        __syncthreads();
    }
}

// ---------------------------------------------------------------------------
// Kernel 5: out = tanh(last 6 elements of final hidden state)
// ---------------------------------------------------------------------------
__global__ void final_kernel(const float* __restrict__ hlast,
                             float* __restrict__ out)
{
    const int j = threadIdx.x;
    if (j < 6) out[j] = tanhf(hlast[(HID - 6) + j]);
}

// ---------------------------------------------------------------------------
// Host-side orchestration (graph-capturable: only async launches on `stream`).
// Input order (flattened setup_inputs dict):
//  0 Is  1 hklss  2 fs
//  3..6   I_params  (W[1,64], b[64], W[64,6], b[6])
//  7..12  hkls_params (Wx[3,256], Wh[64,256], b[256], Wx[64,24], Wh[6,24], b[24])
// 13..16  inp_params (W[12,512], b[512], W[512,768], b[768])
// 17..28  f_params x3 (W[2,512], b[512], W[512,768], b[768])
// 29..46  main_params x3 (Wxg[768,3072], Whg[768,3072], bg[3072],
//                         Wcr[768,768], Whr[768,768], br[768])
// ---------------------------------------------------------------------------
extern "C" void kernel_launch(void* const* d_in, const int* in_sizes, int n_in,
                              void* d_out, int out_size, void* d_ws, size_t ws_size,
                              hipStream_t stream)
{
    (void)in_sizes; (void)n_in; (void)out_size; (void)ws_size;
    const float* const* in = (const float* const*)d_in;

    // ---- workspace layout ----
    // f32 region
    float* G     = (float*)d_ws;                    // 2048*3072
    float* femb  = G + (size_t)P_PEAKS * G4;        // 3*768
    float* hlast = femb + 3 * HID;                  // 768
    // bf16 region (16B-aligned: f32 region is 6294528 floats)
    __bf16* b0     = (__bf16*)(hlast + HID);
    __bf16* inp_bf = b0;                            // 2048*32
    __bf16* W1T    = inp_bf + (size_t)P_PEAKS * 32; // 512*32
    __bf16* X1bf   = W1T + 512 * 32;                // 2048*512
    __bf16* W2T    = X1bf + (size_t)P_PEAKS * 512;  // 768*512
    __bf16* xbfA   = W2T + 768 * 512;               // 2048*768
    __bf16* xbfB   = xbfA + (size_t)P_PEAKS * HID;  // 2048*768
    __bf16* WxgT   = xbfB + (size_t)P_PEAKS * HID;  // 3 * 3072*768

    // ---- weight conversion / transposition (bf16) ----
    conv_wt_kernel<<<(512 * 32 + 255) / 256, 256, 0, stream>>>(
        in[13], W1T, 12, 32, 512);
    conv_wt_kernel<<<(768 * 512 + 255) / 256, 256, 0, stream>>>(
        in[15], W2T, 512, 512, 768);
    for (int l = 0; l < 3; ++l) {
        conv_wt_kernel<<<(G4 * HID + 255) / 256, 256, 0, stream>>>(
            in[29 + 6 * l], WxgT + (size_t)l * G4 * HID, HID, HID, G4);
    }

    // 1) per-peak embeddings -> inp_bf [2048,32] (K-padded bf16)
    embed_kernel<<<P_PEAKS / 8, 256, 0, stream>>>(
        in[0], in[1],
        in[3], in[4], in[5], in[6],
        in[7], in[8], in[9], in[10], in[11], in[12],
        inp_bf);

    // 2) input projection: [2048,12] -> relu -> [2048,512] -> [2048,768]
    // (waves = (M/32) * (N/64); 8 waves per 256-thread block)
    wmma_gemm_kernel<32, true>
        <<<(P_PEAKS / 32) * (512 / 64) / 8, 256, 0, stream>>>(
        inp_bf, W1T, in[14], X1bf, P_PEAKS, 512);
    // second linear has no trailing relu in the chain, but downstream only
    // ever consumes relu(x), so fold the loop's relu here (bf16 out).
    wmma_gemm_kernel<512, true>
        <<<(P_PEAKS / 32) * (HID / 64) / 8, 256, 0, stream>>>(
        X1bf, W2T, in[16], xbfA, P_PEAKS, HID);

    // 3) three main LSTM layers (ping-pong xbfA <-> xbfB)
    __bf16* xin  = xbfA;
    __bf16* xout = xbfB;
    for (int l = 0; l < 3; ++l) {
        const float* fW1 = in[17 + 4 * l + 0];
        const float* fb1 = in[17 + 4 * l + 1];
        const float* fW2 = in[17 + 4 * l + 2];
        const float* fb2 = in[17 + 4 * l + 3];
        const float* Whg = in[29 + 6 * l + 1];
        const float* bg  = in[29 + 6 * l + 2];
        const float* Wcr = in[29 + 6 * l + 3];
        const float* Whr = in[29 + 6 * l + 4];
        const float* br  = in[29 + 6 * l + 5];

        femb_kernel<<<1, 256, 0, stream>>>(in[2], fW1, fb1, fW2, fb2,
                                           femb + l * HID);

        // Gpre = relu(x) @ Wxg + bg : [2048,768] x [768,3072] -> f32
        wmma_gemm_kernel<HID, false>
            <<<(P_PEAKS / 32) * (G4 / 64) / 8, 256, 0, stream>>>(
            xin, WxgT + (size_t)l * G4 * HID, bg, G, P_PEAKS, G4);

        main_lstm_kernel<<<1, 1024, 0, stream>>>(
            G, femb + l * HID, Whg, Wcr, Whr, br, xout, hlast);

        __bf16* t = xin; xin = xout; xout = t;
    }

    // 4) output: tanh of last 6 elements of final hidden state
    final_kernel<<<1, 32, 0, stream>>>(hlast, (float*)d_out);
}
